// MambaBlock_5866925326507
// MI455X (gfx1250) — matrix operand order repaired and checked by reading, not compiled, not run on hardware
//
#include <hip/hip_runtime.h>
#include <hip/hip_bf16.h>
#include <math.h>

// ---------------------------------------------------------------- constants
#define D_MODEL   768
#define D_STATE   16
#define D_CONV    4
#define D_INNER   1536
#define DT_RANK   48
#define B_SZ      2
#define L_SEQ     2048
#define ROWS      (B_SZ * L_SEQ)          // 4096
#define N_XZ      (2 * D_INNER)           // 3072
#define N_XDBL    (DT_RANK + 2 * D_STATE) // 80
#define SCAN_CHUNKS 16
#define CHUNK_L   (L_SEQ / SCAN_CHUNKS)   // 128

// ---------------------------------------------------------------- WMMA types
typedef __attribute__((ext_vector_type(16))) __bf16 v16bf;
typedef __attribute__((ext_vector_type(8)))  float  v8f;
typedef __attribute__((ext_vector_type(4)))  unsigned int uint4v;

union Frag {
    uint4v         q[2];
    unsigned short s[16];
    v16bf          v;
};

__device__ __forceinline__ unsigned short f2bf(float f) {
    unsigned int u = __float_as_uint(f);
    return (unsigned short)((u + 0x7FFFu + ((u >> 16) & 1u)) >> 16);
}

// A fragment from LDS-staged tile [16][K]: M = lane&15; lanes 0-15 carry
// K 0-7 / 16-23, lanes 16-31 carry K 8-15 / 24-31 (ISA 7.12.2).
template <int K>
__device__ __forceinline__ v16bf load_a_frag_lds(const unsigned short* smem,
                                                 int k0, int lane) {
    int m  = lane & 15;
    int kb = (lane < 16) ? 0 : 8;
    const unsigned short* p = smem + m * K + k0;
    Frag r;
    r.q[0] = *(const uint4v*)(p + kb);        // ds_load_b128
    r.q[1] = *(const uint4v*)(p + 16 + kb);
    return r.v;
}

// B fragment: 32x16 bf16, lane = column n; lanes 0-15 hold K 0-7 / 16-23, lanes 16-31
// hold K 8-15 / 24-31. Weight stored row-major [N,K] (== B^T): two 8-half runs.
template <int LD>
__device__ __forceinline__ v16bf load_b_frag(const unsigned short* w,
                                             int n0, int k0, int lane) {
    int n    = lane & 15;
    int koff = (lane < 16) ? 0 : 8;
    const unsigned short* p = w + (size_t)(n0 + n) * LD + k0 + koff;
    Frag r;
    r.q[0] = *(const uint4v*)(p);
    r.q[1] = *(const uint4v*)(p + 16);
    return r.v;
}

// A fragment from an f32 source with on-the-fly bf16 convert and compile-time
// K clamp (for the K=48 dt_proj GEMM padded to K=64).
template <int LD, int KMAX>
__device__ __forceinline__ v16bf load_a_frag_f32(const float* src,
                                                 int row0, int k0, int lane) {
    int m  = lane & 15;
    int kb = (lane < 16) ? 0 : 8;
    const float* p = src + (size_t)(row0 + m) * LD;
    Frag r;
#pragma unroll
    for (int j = 0; j < 8; ++j) {
        int k = k0 + kb + j;
        r.s[j] = (k < KMAX) ? f2bf(p[k]) : (unsigned short)0;
    }
#pragma unroll
    for (int j = 0; j < 8; ++j) {
        int k = k0 + 16 + kb + j;
        r.s[8 + j] = (k < KMAX) ? f2bf(p[k]) : (unsigned short)0;
    }
    return r.v;
}

template <int NT>
__device__ __forceinline__ void zero_acc(v8f acc[NT]) {
#pragma unroll
    for (int t = 0; t < NT; ++t)
#pragma unroll
        for (int i = 0; i < 8; ++i) acc[t][i] = 0.f;
}

// Cooperative async stage of the block's shared A tile (16 rows x K bf16,
// contiguous in global) into LDS via the gfx1250 ASYNCcnt path.
// Low 32 bits of a flat LDS pointer are the LDS byte offset (aperture is in
// addr[63:32]), so truncation gives the VGPR LDS address the instruction wants.
template <int K>
__device__ __forceinline__ void stage_a_tile(const unsigned short* __restrict__ gA,
                                             int m0, unsigned short* smem,
                                             int tid, int nthreads) {
    const unsigned short* src = gA + (size_t)m0 * K;
    constexpr int CHUNKS = 16 * K / 8;     // 16-byte chunks
    for (int i = tid; i < CHUNKS; i += nthreads) {
        unsigned long long g = (unsigned long long)(uintptr_t)(src + i * 8);
        unsigned l = (unsigned)(uintptr_t)(smem + i * 8);
        asm volatile("global_load_async_to_lds_b128 %0, %1, off"
                     :: "v"(l), "v"(g) : "memory");
    }
    asm volatile("s_wait_asynccnt 0" ::: "memory");
    __syncthreads();
}

// Mainloop: one wave computes a 16 x (16*NT) tile; A from LDS, B from global.
// All bounds compile-time, no control flow inside -> EXEC all-1s at WMMA.
template <int K, int NT>
__device__ __forceinline__ void gemm_loop_lds(const unsigned short* smemA,
                                              const unsigned short* __restrict__ W,
                                              int n0base, int lane, v8f acc[NT]) {
#pragma unroll 2
    for (int k0 = 0; k0 < K; k0 += 32) {
        v16bf af = load_a_frag_lds<K>(smemA, k0, lane);
#pragma unroll
        for (int t = 0; t < NT; ++t) {
            v16bf bf = load_b_frag<K>(W, n0base + t * 16, k0, lane);
            acc[t] = __builtin_amdgcn_wmma_f32_16x16x32_bf16(
                false, af, false, bf, (short)0, acc[t], false, false);
        }
    }
}

// ---------------------------------------------------------------- converts
__global__ void k_cvt_f32_bf16(const float* __restrict__ in,
                               unsigned short* __restrict__ out, int n) {
    int i = blockIdx.x * blockDim.x + threadIdx.x;
    if (i < n) out[i] = f2bf(in[i]);
}

// dt_proj_w [1536,48] -> bf16 padded to [1536,64] so K is a multiple of 32.
__global__ void k_cvt_pad_dtw(const float* __restrict__ in,
                              unsigned short* __restrict__ out) {
    int i = blockIdx.x * blockDim.x + threadIdx.x;
    if (i >= D_INNER * 64) return;
    int r = i >> 6, c = i & 63;
    out[i] = (c < DT_RANK) ? f2bf(in[r * DT_RANK + c]) : (unsigned short)0;
}

// ---------------------------------------------------------------- layernorm
// One wave32 per row (768 elems -> 24/lane), shuffle reduction, bf16 output.
__global__ void k_layernorm(const float* __restrict__ x,
                            const float* __restrict__ g,
                            const float* __restrict__ be,
                            unsigned short* __restrict__ xn_bf) {
    int wave = threadIdx.x >> 5, lane = threadIdx.x & 31;
    int row = blockIdx.x * (blockDim.x >> 5) + wave;
    if (row >= ROWS) return;
    const float* xr = x + (size_t)row * D_MODEL;
    float s = 0.f;
    for (int i = lane; i < D_MODEL; i += 32) s += xr[i];
#pragma unroll
    for (int o = 16; o > 0; o >>= 1) s += __shfl_xor(s, o, 32);
    float mu = s * (1.f / D_MODEL);
    float v = 0.f;
    for (int i = lane; i < D_MODEL; i += 32) { float d = xr[i] - mu; v += d * d; }
#pragma unroll
    for (int o = 16; o > 0; o >>= 1) v += __shfl_xor(v, o, 32);
    float rstd = rsqrtf(v * (1.f / D_MODEL) + 1e-5f);
    for (int i = lane; i < D_MODEL; i += 32)
        xn_bf[(size_t)row * D_MODEL + i] = f2bf((xr[i] - mu) * rstd * g[i] + be[i]);
}

// ---------------------------------------------------------------- GEMMs
// in_proj: 4096 x 3072 x 768. 4 waves/block, 64 cols/wave; shared A tile in LDS.
__global__ void k_gemm_inproj(const unsigned short* __restrict__ A,
                              const unsigned short* __restrict__ W,
                              float* __restrict__ C) {
    constexpr int N = N_XZ, K = D_MODEL;
    __shared__ unsigned short smA[16 * K];           // 24 KB
    int lane = threadIdx.x & 31, wave = threadIdx.x >> 5;
    int m0 = blockIdx.y * 16;
    int n0base = (blockIdx.x * 4 + wave) * 64;
    stage_a_tile<K>(A, m0, smA, threadIdx.x, 128);
    v8f acc[4]; zero_acc<4>(acc);
    gemm_loop_lds<K, 4>(smA, W, n0base, lane, acc);
    int col = lane & 15, ro = (lane >> 4) * 8;
#pragma unroll
    for (int t = 0; t < 4; ++t)
#pragma unroll
        for (int r = 0; r < 8; ++r)
            C[(size_t)(m0 + ro + r) * N + n0base + t * 16 + col] = acc[t][r];
}

// x_proj: 4096 x 80 x 1536. 5 waves/block (160 threads), 1 tile/wave; the
// grid exactly covers N=80, so no guards and full EXEC everywhere.
__global__ void k_gemm_xproj(const unsigned short* __restrict__ A,
                             const unsigned short* __restrict__ W,
                             float* __restrict__ C) {
    constexpr int N = N_XDBL, K = D_INNER;
    __shared__ unsigned short smA[16 * K];           // 48 KB
    int lane = threadIdx.x & 31, wave = threadIdx.x >> 5;
    int m0 = blockIdx.y * 16;
    int n0 = wave * 16;     // 0..64
    stage_a_tile<K>(A, m0, smA, threadIdx.x, 160);
    v8f acc[1]; zero_acc<1>(acc);
    gemm_loop_lds<K, 1>(smA, W, n0, lane, acc);
    int col = lane & 15, ro = (lane >> 4) * 8;
#pragma unroll
    for (int r = 0; r < 8; ++r)
        C[(size_t)(m0 + ro + r) * N + n0 + col] = acc[0][r];
}

// dt = softplus(dt_lo @ Wdt^T + b); dt_lo = cols [0,48) of xdbl (ld=80),
// K padded to 64 at compile time. Tiny K -> direct loads, no staging.
__global__ void k_gemm_dt(const float* __restrict__ xdbl,
                          const unsigned short* __restrict__ Wdt,
                          const float* __restrict__ dtb,
                          float* __restrict__ dtout) {
    constexpr int N = D_INNER;
    int lane = threadIdx.x & 31, wave = threadIdx.x >> 5;
    int m0 = blockIdx.y * 16;
    int n0base = (blockIdx.x * 4 + wave) * 64;
    v8f acc[4]; zero_acc<4>(acc);
#pragma unroll
    for (int k0 = 0; k0 < 64; k0 += 32) {
        v16bf af = load_a_frag_f32<N_XDBL, DT_RANK>(xdbl, m0, k0, lane);
#pragma unroll
        for (int t = 0; t < 4; ++t) {
            v16bf bf = load_b_frag<64>(Wdt, n0base + t * 16, k0, lane);
            acc[t] = __builtin_amdgcn_wmma_f32_16x16x32_bf16(
                false, af, false, bf, (short)0, acc[t], false, false);
        }
    }
    int col = lane & 15, ro = (lane >> 4) * 8;
#pragma unroll
    for (int t = 0; t < 4; ++t) {
        float bias = dtb[n0base + t * 16 + col];
#pragma unroll
        for (int r = 0; r < 8; ++r) {
            float v = acc[t][r] + bias;
            float sp = (v > 20.f) ? v : log1pf(__expf(v));
            dtout[(size_t)(m0 + ro + r) * N + n0base + t * 16 + col] = sp;
        }
    }
}

// out = x + yg @ Wout^T (residual fused into epilogue). 4096 x 768 x 1536.
__global__ void k_gemm_outproj(const unsigned short* __restrict__ A,
                               const unsigned short* __restrict__ W,
                               const float* __restrict__ xres,
                               float* __restrict__ C) {
    constexpr int N = D_MODEL, K = D_INNER;
    __shared__ unsigned short smA[16 * K];           // 48 KB
    int lane = threadIdx.x & 31, wave = threadIdx.x >> 5;
    int m0 = blockIdx.y * 16;
    int n0base = (blockIdx.x * 4 + wave) * 64;
    stage_a_tile<K>(A, m0, smA, threadIdx.x, 128);
    v8f acc[4]; zero_acc<4>(acc);
    gemm_loop_lds<K, 4>(smA, W, n0base, lane, acc);
    int col = lane & 15, ro = (lane >> 4) * 8;
#pragma unroll
    for (int t = 0; t < 4; ++t)
#pragma unroll
        for (int r = 0; r < 8; ++r) {
            size_t idx = (size_t)(m0 + ro + r) * N + n0base + t * 16 + col;
            C[idx] = xres[idx] + acc[t][r];
        }
}

// ---------------------------------------------------------------- conv + SiLU
__global__ void k_conv_silu(const float* __restrict__ xz,
                            const float* __restrict__ cw,
                            const float* __restrict__ cb,
                            float* __restrict__ uf,
                            unsigned short* __restrict__ ub) {
    int idx = blockIdx.x * blockDim.x + threadIdx.x;
    if (idx >= ROWS * D_INNER) return;
    int d = idx % D_INNER;
    int l = (idx / D_INNER) % L_SEQ;
    int b = idx / (D_INNER * L_SEQ);
    float acc = cb[d];
#pragma unroll
    for (int j = 0; j < D_CONV; ++j) {
        int t = l - (D_CONV - 1) + j;
        if (t >= 0)
            acc += cw[d * D_CONV + j] * xz[(size_t)(b * L_SEQ + t) * N_XZ + d];
    }
    float s = acc / (1.f + __expf(-acc));   // SiLU
    uf[idx] = s;
    ub[idx] = f2bf(s);
}

// ---------------------------------------------------------------- chunked scan
// Phase 1: per chunk, compute partial state hf (h0=0) and decay product P.
__global__ void k_scan1(const float* __restrict__ u, const float* __restrict__ dtv,
                        const float* __restrict__ xdbl, const float* __restrict__ A_log,
                        float* __restrict__ ch_h, float* __restrict__ ch_P) {
    int d = blockIdx.x * blockDim.x + threadIdx.x;
    int b = blockIdx.y, c = blockIdx.z;
    if (d >= D_INNER) return;
    float A[D_STATE], h[D_STATE], P[D_STATE];
#pragma unroll
    for (int n = 0; n < D_STATE; ++n) {
        A[n] = -__expf(A_log[d * D_STATE + n]);
        h[n] = 0.f; P[n] = 1.f;
    }
    int l0 = c * CHUNK_L;
    for (int l = l0; l < l0 + CHUNK_L; ++l) {
        int row = b * L_SEQ + l;
        float dtt = dtv[(size_t)row * D_INNER + d];
        float dtu = dtt * u[(size_t)row * D_INNER + d];
        const float* Bt = xdbl + (size_t)row * N_XDBL + DT_RANK;
#pragma unroll
        for (int n = 0; n < D_STATE; ++n) {
            float dA = __expf(dtt * A[n]);
            P[n] *= dA;
            h[n] = dA * h[n] + dtu * Bt[n];
        }
    }
    size_t base = ((size_t)(b * SCAN_CHUNKS + c) * D_INNER + d) * D_STATE;
#pragma unroll
    for (int n = 0; n < D_STATE; ++n) { ch_h[base + n] = h[n]; ch_P[base + n] = P[n]; }
}

// Phase 2: sequential prefix over the 16 chunks; rewrite ch_h[c] = h_init[c].
__global__ void k_scanfix(float* __restrict__ ch_h, const float* __restrict__ ch_P) {
    int d = blockIdx.x * blockDim.x + threadIdx.x;
    int b = blockIdx.y;
    if (d >= D_INNER) return;
    float h[D_STATE];
#pragma unroll
    for (int n = 0; n < D_STATE; ++n) h[n] = 0.f;
    for (int c = 0; c < SCAN_CHUNKS; ++c) {
        size_t base = ((size_t)(b * SCAN_CHUNKS + c) * D_INNER + d) * D_STATE;
#pragma unroll
        for (int n = 0; n < D_STATE; ++n) {
            float hf = ch_h[base + n];
            float nh = ch_P[base + n] * h[n] + hf;
            ch_h[base + n] = h[n];          // becomes the chunk's initial state
            h[n] = nh;
        }
    }
}

// Phase 3: replay each chunk from corrected h_init, emit y, gate, bf16 output.
__global__ void k_scan2(const float* __restrict__ u, const float* __restrict__ dtv,
                        const float* __restrict__ xdbl, const float* __restrict__ A_log,
                        const float* __restrict__ Dp, const float* __restrict__ xz,
                        const float* __restrict__ ch_h, unsigned short* __restrict__ yg) {
    int d = blockIdx.x * blockDim.x + threadIdx.x;
    int b = blockIdx.y, c = blockIdx.z;
    if (d >= D_INNER) return;
    float A[D_STATE], h[D_STATE];
    size_t base = ((size_t)(b * SCAN_CHUNKS + c) * D_INNER + d) * D_STATE;
#pragma unroll
    for (int n = 0; n < D_STATE; ++n) {
        A[n] = -__expf(A_log[d * D_STATE + n]);
        h[n] = ch_h[base + n];
    }
    float Dd = Dp[d];
    int l0 = c * CHUNK_L;
    for (int l = l0; l < l0 + CHUNK_L; ++l) {
        int row = b * L_SEQ + l;
        float ut  = u[(size_t)row * D_INNER + d];
        float dtt = dtv[(size_t)row * D_INNER + d];
        float dtu = dtt * ut;
        const float* Bt = xdbl + (size_t)row * N_XDBL + DT_RANK;
        const float* Ct = xdbl + (size_t)row * N_XDBL + DT_RANK + D_STATE;
        float y = 0.f;
#pragma unroll
        for (int n = 0; n < D_STATE; ++n) {
            float dA = __expf(dtt * A[n]);
            h[n] = dA * h[n] + dtu * Bt[n];
            y += Ct[n] * h[n];
        }
        y += Dd * ut;
        float z = xz[(size_t)row * N_XZ + D_INNER + d];
        float sz = z / (1.f + __expf(-z));
        yg[(size_t)row * D_INNER + d] = f2bf(y * sz);
    }
}

// ---------------------------------------------------------------- host launch
extern "C" void kernel_launch(void* const* d_in, const int* in_sizes, int n_in,
                              void* d_out, int out_size, void* d_ws, size_t ws_size,
                              hipStream_t stream) {
    (void)in_sizes; (void)n_in; (void)out_size; (void)ws_size;
    const float* x      = (const float*)d_in[0];
    const float* ln_g   = (const float*)d_in[1];
    const float* ln_b   = (const float*)d_in[2];
    const float* w_in   = (const float*)d_in[3];
    const float* conv_w = (const float*)d_in[4];
    const float* conv_b = (const float*)d_in[5];
    const float* w_xp   = (const float*)d_in[6];
    const float* w_dt   = (const float*)d_in[7];
    const float* dt_b   = (const float*)d_in[8];
    const float* A_log  = (const float*)d_in[9];
    const float* Dp     = (const float*)d_in[10];
    const float* w_out  = (const float*)d_in[11];
    float* out = (float*)d_out;

    char* ws = (char*)d_ws;
    size_t off = 0;
    auto wsalloc = [&](size_t bytes) -> void* {
        void* p = ws + off;
        off += (bytes + 255) & ~(size_t)255;
        return p;
    };
    unsigned short* xn_bf   = (unsigned short*)wsalloc((size_t)ROWS * D_MODEL * 2);
    unsigned short* w_in_bf = (unsigned short*)wsalloc((size_t)N_XZ * D_MODEL * 2);
    unsigned short* w_xp_bf = (unsigned short*)wsalloc((size_t)N_XDBL * D_INNER * 2);
    unsigned short* w_dt_bf = (unsigned short*)wsalloc((size_t)D_INNER * 64 * 2);
    unsigned short* w_out_bf= (unsigned short*)wsalloc((size_t)D_MODEL * D_INNER * 2);
    float*          xz      = (float*)wsalloc((size_t)ROWS * N_XZ * 4);
    float*          u_f32   = (float*)wsalloc((size_t)ROWS * D_INNER * 4);
    unsigned short* u_bf    = (unsigned short*)wsalloc((size_t)ROWS * D_INNER * 2);
    float*          xdbl    = (float*)wsalloc((size_t)ROWS * N_XDBL * 4);
    float*          dtv     = (float*)wsalloc((size_t)ROWS * D_INNER * 4);
    unsigned short* yg_bf   = (unsigned short*)wsalloc((size_t)ROWS * D_INNER * 2);
    float*          ch_h    = (float*)wsalloc((size_t)B_SZ * SCAN_CHUNKS * D_INNER * D_STATE * 4);
    float*          ch_P    = (float*)wsalloc((size_t)B_SZ * SCAN_CHUNKS * D_INNER * D_STATE * 4);

    // weight converts
    {
        int n = N_XZ * D_MODEL;
        k_cvt_f32_bf16<<<(n + 255) / 256, 256, 0, stream>>>(w_in, w_in_bf, n);
        n = N_XDBL * D_INNER;
        k_cvt_f32_bf16<<<(n + 255) / 256, 256, 0, stream>>>(w_xp, w_xp_bf, n);
        n = D_MODEL * D_INNER;
        k_cvt_f32_bf16<<<(n + 255) / 256, 256, 0, stream>>>(w_out, w_out_bf, n);
        n = D_INNER * 64;
        k_cvt_pad_dtw<<<(n + 255) / 256, 256, 0, stream>>>(w_dt, w_dt_bf);
    }

    // layernorm -> bf16 (wave per row, 8 rows/block)
    k_layernorm<<<ROWS / 8, 256, 0, stream>>>(x, ln_g, ln_b, xn_bf);

    // in_proj: 4096x3072x768, 64 cols/wave, 4 waves/block, A tile in LDS
    k_gemm_inproj<<<dim3(N_XZ / 256, ROWS / 16), 128, 0, stream>>>(xn_bf, w_in_bf, xz);

    // depthwise causal conv + SiLU
    {
        int n = ROWS * D_INNER;
        k_conv_silu<<<(n + 255) / 256, 256, 0, stream>>>(xz, conv_w, conv_b, u_f32, u_bf);
    }

    // x_proj: 4096x80x1536 -- 5 waves/block, one 16-col tile each
    k_gemm_xproj<<<dim3(1, ROWS / 16), 160, 0, stream>>>(u_bf, w_xp_bf, xdbl);

    // dt_proj + softplus: 4096x1536x48 (K padded to 64)
    k_gemm_dt<<<dim3(D_INNER / 256, ROWS / 16), 128, 0, stream>>>(xdbl, w_dt_bf, dt_b, dtv);

    // chunked selective scan (16x parallel linear-recurrence scan)
    k_scan1<<<dim3(D_INNER / 256, B_SZ, SCAN_CHUNKS), 256, 0, stream>>>(
        u_f32, dtv, xdbl, A_log, ch_h, ch_P);
    k_scanfix<<<dim3(D_INNER / 256, B_SZ), 256, 0, stream>>>(ch_h, ch_P);
    k_scan2<<<dim3(D_INNER / 256, B_SZ, SCAN_CHUNKS), 256, 0, stream>>>(
        u_f32, dtv, xdbl, A_log, Dp, xz, ch_h, yg_bf);

    // out_proj + residual: 4096x768x1536
    k_gemm_outproj<<<dim3(D_MODEL / 256, ROWS / 16), 128, 0, stream>>>(
        yg_bf, w_out_bf, x, out);
}
